// cross_context_75333726372152
// MI455X (gfx1250) — matrix operand order repaired and checked by reading, not compiled, not run on hardware
//
#include <hip/hip_runtime.h>

// MI455X / gfx1250, wave32. Pipeline for VN cross-attention reference.
// B=2, C=32, N=768, K=16.

typedef __attribute__((ext_vector_type(16))) _Float16 v16h;
typedef __attribute__((ext_vector_type(8)))  float    v8f;

#define BATCH 2
#define NPTS  768
#define CIN   32
#define KNN   16
#define FDIM  96           // CIN*3
#define EPS_VN   1e-6f
#define EPS_NRM  1e-12f
#define NEG_SLOPE 0.2f

// ---------------------------------------------------------------- sq[b,n]
__global__ __launch_bounds__(128) void vn_sq_kernel(const float* __restrict__ y,
                                                    float* __restrict__ sq) {
    int i = blockIdx.x * blockDim.x + threadIdx.x;     // b*768+n
    if (i >= BATCH * NPTS) return;
    int b = i / NPTS, n = i % NPTS;
    const float* yf = y + (size_t)b * FDIM * NPTS;     // [96,768]
    float s = 0.f;
    for (int f = 0; f < FDIM; ++f) { float v = yf[f * NPTS + n]; s += v * v; }
    sq[i] = s;
}

// --------------------------------------------------- Gram via WMMA f16 16x16x32
// neg_dist[b,n,m] = 2*G[n,m] - sq[n] - sq[m],  G = yf^T yf  (yf: [96,768])
__global__ __launch_bounds__(32) void vn_gram_kernel(const float* __restrict__ y,
                                                     const float* __restrict__ sq,
                                                     float* __restrict__ nd) {
    const int nb = blockIdx.x * 16;        // row tile (M)
    const int mb = blockIdx.y * 16;        // col tile (N)
    const int b  = blockIdx.z;
    const float* yf = y + (size_t)b * FDIM * NPTS;
    const int lane = threadIdx.x;          // 0..31, EXEC all ones
    const int h    = lane >> 4;            // lane half
    const int l16  = lane & 15;

    v8f acc = {};
    for (int fb = 0; fb < FDIM; fb += 32) {
        v16h av, bv;
        // A (16x32, 16-bit layout): lane -> M=l16; element e=(v,p):
        //   v<4 : K = 8*h + 2*v + p ;  v>=4 : K = 16 + 8*h + 2*(v-4) + p
        // B (32x16, 16-bit layout): lane -> Ncol=l16; element e: e<8 : K = 8*h + e ;  e>=8 : K = 16 + 8*h + (e-8)
        #pragma unroll
        for (int e = 0; e < 16; ++e) {
            int v = e >> 1, p = e & 1;
            int kA = (v < 4) ? (8 * h + 2 * v + p) : (16 + 8 * h + 2 * (v - 4) + p);
            av[e] = (_Float16)yf[(fb + kA) * NPTS + (nb + l16)];   // A[M,K]=yf[K,M]
            int kB = (e < 8) ? (8 * h + e) : (16 + 8 * h + (e - 8));
            bv[e] = (_Float16)yf[(fb + kB) * NPTS + (mb + l16)];   // B[K,N]=yf[K,N]
        }
        acc = __builtin_amdgcn_wmma_f32_16x16x32_f16(
            /*neg_a=*/false, av, /*neg_b=*/false, bv,
            /*c_mod=*/(short)0, acc, /*reuse_a=*/false, /*reuse_b=*/false);
    }
    // D layout: vgpr r, lane -> M = r + 8*h, N = l16
    const float sqm = sq[b * NPTS + (mb + l16)];
    #pragma unroll
    for (int r = 0; r < 8; ++r) {
        int n = nb + r + 8 * h;
        int m = mb + l16;
        nd[((size_t)b * NPTS + n) * NPTS + m] = 2.f * acc[r] - sq[b * NPTS + n] - sqm;
    }
}

// ---------------------------------------------------------------- top-16 / row
__global__ __launch_bounds__(64) void vn_topk_kernel(const float* __restrict__ nd,
                                                     int* __restrict__ idx) {
    int row = blockIdx.x * blockDim.x + threadIdx.x;   // b*768+n
    if (row >= BATCH * NPTS) return;
    const float* r = nd + (size_t)row * NPTS;
    float vals[KNN]; int ids[KNN];
    #pragma unroll
    for (int i = 0; i < KNN; ++i) { vals[i] = -1e30f; ids[i] = 0; }
    for (int m = 0; m < NPTS; ++m) {
        float v = r[m];
        if (v > vals[KNN - 1]) {
            int pos = KNN - 1;
            while (pos > 0 && vals[pos - 1] < v) {
                vals[pos] = vals[pos - 1]; ids[pos] = ids[pos - 1]; --pos;
            }
            vals[pos] = v; ids[pos] = m;
        }
    }
    #pragma unroll
    for (int i = 0; i < KNN; ++i) idx[row * KNN + i] = ids[i];
}

// ------------------------------------------------ Qx: VN linear-leaky + normalize
// one wave (32 threads = 32 out-channels) per (b,n)
__global__ __launch_bounds__(32) void vn_qx_kernel(const float* __restrict__ x,
                                                   const float* __restrict__ Wf,
                                                   const float* __restrict__ Wd,
                                                   float* __restrict__ Qx) {
    __shared__ float xs[CIN][3];
    __shared__ float red[32];
    const int n = blockIdx.x, b = blockIdx.y, co = threadIdx.x;
    for (int t = co; t < FDIM; t += 32) {
        int ci = t / 3, d = t % 3;
        xs[ci][d] = x[(((size_t)b * CIN + ci) * 3 + d) * NPTS + n];
    }
    __syncthreads();
    float p[3] = {0, 0, 0}, dd[3] = {0, 0, 0};
    for (int ci = 0; ci < CIN; ++ci) {
        float wf = Wf[co * CIN + ci], wd = Wd[co * CIN + ci];
        #pragma unroll
        for (int d = 0; d < 3; ++d) { p[d] += wf * xs[ci][d]; dd[d] += wd * xs[ci][d]; }
    }
    float dot = p[0]*dd[0] + p[1]*dd[1] + p[2]*dd[2];
    float dsq = dd[0]*dd[0] + dd[1]*dd[1] + dd[2]*dd[2];
    float q[3];
    #pragma unroll
    for (int d = 0; d < 3; ++d) {
        float neg = p[d] - (dot / (dsq + EPS_VN)) * dd[d];
        q[d] = NEG_SLOPE * p[d] + (1.f - NEG_SLOPE) * ((dot >= 0.f) ? p[d] : neg);
    }
    float nrm2 = q[0]*q[0] + q[1]*q[1] + q[2]*q[2];
    red[co] = nrm2;
    __syncthreads();
    float s = 0.f;
    for (int i = 0; i < 32; ++i) s += red[i];
    float nrm = sqrtf(nrm2);
    float fac = (nrm / fmaxf(sqrtf(s), EPS_NRM)) / fmaxf(nrm, EPS_NRM);
    #pragma unroll
    for (int d = 0; d < 3; ++d)
        Qx[(((size_t)b * CIN + co) * 3 + d) * NPTS + n] = q[d] * fac;
}

// ------------- graph feature -> K/V VN-linear-leaky (+normalize K) -> sums over k
// one wave (32 threads = 32 out-channels) per (b,n)
__global__ __launch_bounds__(32) void vn_kv_kernel(const float* __restrict__ y,
                                                   const int* __restrict__ idx,
                                                   const float* __restrict__ Wkf,
                                                   const float* __restrict__ Wkd,
                                                   const float* __restrict__ Wvf,
                                                   const float* __restrict__ Wvd,
                                                   float* __restrict__ Ksum,
                                                   float* __restrict__ Vsum) {
    __shared__ float ctr[FDIM];   // center slice y[b,:,:,n]  (ci*3+d)
    __shared__ float nb[FDIM];    // neighbor slice
    __shared__ float red[32];
    const int n = blockIdx.x, b = blockIdx.y, co = threadIdx.x;
    const float* yb = y + (size_t)b * FDIM * NPTS;
    for (int t = co; t < FDIM; t += 32) ctr[t] = yb[t * NPTS + n];

    float ks[3] = {0, 0, 0}, vs[3] = {0, 0, 0};
    for (int j = 0; j < KNN; ++j) {
        __syncthreads();                                 // prior nb/red reads done
        int m = idx[(b * NPTS + n) * KNN + j];
        for (int t = co; t < FDIM; t += 32) nb[t] = yb[t * NPTS + m];
        __syncthreads();                                 // nb ready

        float pk[3]={0,0,0}, dk[3]={0,0,0}, pv[3]={0,0,0}, dv[3]={0,0,0};
        for (int ci = 0; ci < CIN; ++ci) {
            float wkf0 = Wkf[co*64 + ci],      wkd0 = Wkd[co*64 + ci];
            float wvf0 = Wvf[co*64 + ci],      wvd0 = Wvd[co*64 + ci];
            float wkf1 = Wkf[co*64 + 32 + ci], wkd1 = Wkd[co*64 + 32 + ci];
            float wvf1 = Wvf[co*64 + 32 + ci], wvd1 = Wvd[co*64 + 32 + ci];
            #pragma unroll
            for (int d = 0; d < 3; ++d) {
                float gc = ctr[ci*3 + d];
                float ge = nb[ci*3 + d] - gc;            // edge feature
                pk[d] += wkf0 * ge + wkf1 * gc;
                dk[d] += wkd0 * ge + wkd1 * gc;
                pv[d] += wvf0 * ge + wvf1 * gc;
                dv[d] += wvd0 * ge + wvd1 * gc;
            }
        }
        // leaky (K)
        float dotk = pk[0]*dk[0] + pk[1]*dk[1] + pk[2]*dk[2];
        float dsqk = dk[0]*dk[0] + dk[1]*dk[1] + dk[2]*dk[2];
        float kq[3];
        #pragma unroll
        for (int d = 0; d < 3; ++d) {
            float neg = pk[d] - (dotk / (dsqk + EPS_VN)) * dk[d];
            kq[d] = NEG_SLOPE * pk[d] + (1.f - NEG_SLOPE) * ((dotk >= 0.f) ? pk[d] : neg);
        }
        // leaky (V) -> accumulate
        float dotv = pv[0]*dv[0] + pv[1]*dv[1] + pv[2]*dv[2];
        float dsqv = dv[0]*dv[0] + dv[1]*dv[1] + dv[2]*dv[2];
        #pragma unroll
        for (int d = 0; d < 3; ++d) {
            float neg = pv[d] - (dotv / (dsqv + EPS_VN)) * dv[d];
            vs[d] += NEG_SLOPE * pv[d] + (1.f - NEG_SLOPE) * ((dotv >= 0.f) ? pv[d] : neg);
        }
        // channel-equi normalize K over the 32 channels (this wave)
        float nrm2 = kq[0]*kq[0] + kq[1]*kq[1] + kq[2]*kq[2];
        red[co] = nrm2;
        __syncthreads();
        float s = 0.f;
        for (int i = 0; i < 32; ++i) s += red[i];
        float nrm = sqrtf(nrm2);
        float fac = (nrm / fmaxf(sqrtf(s), EPS_NRM)) / fmaxf(nrm, EPS_NRM);
        #pragma unroll
        for (int d = 0; d < 3; ++d) ks[d] += kq[d] * fac;
    }
    #pragma unroll
    for (int d = 0; d < 3; ++d) {
        size_t o = (((size_t)b * CIN + co) * 3 + d) * NPTS + n;
        Ksum[o] = ks[d];
        Vsum[o] = vs[d];
    }
}

// ---------------- attention: rank-1 logits -> softmax -> weighted Vsum + residual
// one block per (b,c,d) channel; s,V staged in LDS
__global__ __launch_bounds__(256) void vn_attn_kernel(const float* __restrict__ x,
                                                      const float* __restrict__ Qx,
                                                      const float* __restrict__ Ksum,
                                                      const float* __restrict__ Vsum,
                                                      float* __restrict__ out) {
    __shared__ float sS[NPTS], sV[NPTS], red[256];
    const float scale = 0.1020620726159658f;      // 1/sqrt(3*32)
    const int ch  = blockIdx.x;                   // (b*32+c)*3+d, 0..191
    const int tid = threadIdx.x;
    const size_t base = (size_t)ch * NPTS;

    float lmax = -1e30f, lmin = 1e30f;
    for (int m = tid; m < NPTS; m += 256) {
        float sv = Ksum[base + m] * scale;
        sS[m] = sv; sV[m] = Vsum[base + m];
        lmax = fmaxf(lmax, sv); lmin = fminf(lmin, sv);
    }
    red[tid] = lmax; __syncthreads();
    for (int o = 128; o > 0; o >>= 1) {
        if (tid < o) red[tid] = fmaxf(red[tid], red[tid + o]);
        __syncthreads();
    }
    float chmax = red[0]; __syncthreads();
    red[tid] = lmin; __syncthreads();
    for (int o = 128; o > 0; o >>= 1) {
        if (tid < o) red[tid] = fminf(red[tid], red[tid + o]);
        __syncthreads();
    }
    float chmin = red[0]; __syncthreads();

    for (int n = tid; n < NPTS; n += 256) {
        float q = Qx[base + n];
        float rmax = (q >= 0.f) ? q * chmax : q * chmin;   // max_m of q*s_m
        float Z = 0.f, W = 0.f;
        for (int m = 0; m < NPTS; ++m) {
            float e = __expf(q * sS[m] - rmax);
            Z += e; W += e * sV[m];
        }
        out[base + n] = x[base + n] + W / Z;
    }
}

extern "C" void kernel_launch(void* const* d_in, const int* in_sizes, int n_in,
                              void* d_out, int out_size, void* d_ws, size_t ws_size,
                              hipStream_t stream) {
    (void)in_sizes; (void)n_in; (void)out_size; (void)ws_size;
    const float* x   = (const float*)d_in[0];
    const float* y   = (const float*)d_in[1];
    const float* Wqf = (const float*)d_in[2];
    const float* Wqd = (const float*)d_in[3];
    const float* Wkf = (const float*)d_in[4];
    const float* Wkd = (const float*)d_in[5];
    const float* Wvf = (const float*)d_in[6];
    const float* Wvd = (const float*)d_in[7];
    float* out = (float*)d_out;

    // workspace layout (floats): sq | nd | Qx | Ksum | Vsum | idx(int)
    float* ws   = (float*)d_ws;
    float* sq   = ws;                                   // 1536
    float* nd   = sq + BATCH * NPTS;                    // 1,179,648
    float* Qx   = nd + (size_t)BATCH * NPTS * NPTS;     // 147,456
    float* Ks   = Qx + (size_t)BATCH * CIN * 3 * NPTS;  // 147,456
    float* Vs   = Ks + (size_t)BATCH * CIN * 3 * NPTS;  // 147,456
    int*   idx  = (int*)(Vs + (size_t)BATCH * CIN * 3 * NPTS);   // 24,576 ints

    vn_sq_kernel  <<<dim3((BATCH * NPTS + 127) / 128), dim3(128), 0, stream>>>(y, sq);
    vn_gram_kernel<<<dim3(NPTS / 16, NPTS / 16, BATCH), dim3(32), 0, stream>>>(y, sq, nd);
    vn_topk_kernel<<<dim3((BATCH * NPTS + 63) / 64), dim3(64), 0, stream>>>(nd, idx);
    vn_qx_kernel  <<<dim3(NPTS, BATCH), dim3(32), 0, stream>>>(x, Wqf, Wqd, Qx);
    vn_kv_kernel  <<<dim3(NPTS, BATCH), dim3(32), 0, stream>>>(y, idx, Wkf, Wkd, Wvf, Wvd, Ks, Vs);
    vn_attn_kernel<<<dim3(BATCH * CIN * 3), dim3(256), 0, stream>>>(x, Qx, Ks, Vs, out);
}
